// self_attention_26877905338955
// MI455X (gfx1250) — matrix operand order repaired and checked by reading, not compiled, not run on hardware
//
#include <hip/hip_runtime.h>
#include <stdint.h>

// ---------------------------------------------------------------------------
// Fused self-attention for MI455X (gfx1250, wave32, WMMA + async LDS copy).
//   B=2, S=2048, E=1024, H=16, D=64.
//   softmax is over the HEAD axis (replicating the reference), which makes the
//   attention perfectly streamable over k with no running-max rescale.
// ---------------------------------------------------------------------------

typedef __bf16 bf16;
typedef __attribute__((ext_vector_type(16))) __bf16 bf16x16;
typedef __attribute__((ext_vector_type(8)))  float  f32x8;

#define NB 2
#define NS 2048
#define NE 1024
#define NH 16
#define ND 64
#define NM (NB * NS)      // 4096 rows of x
#define N_QKV (3 * NE)    // 3072

__device__ __forceinline__ int lane_id() { return (int)(threadIdx.x & 31); }
__device__ __forceinline__ int wave_id() { return (int)(threadIdx.x >> 5); }

// ---- CDNA5 async global->LDS copy (ASYNCcnt-tracked, no VGPR data) ---------

// One 16B chunk per lane, memory -> LDS directly. LDS address is the low 32
// bits of the generic pointer (ISA 10.2: LDS aperture addr[31:0] = offset).
__device__ __forceinline__ void async_copy_b128(void* lds_ptr, const void* gptr) {
  const uint32_t laddr = (uint32_t)(uintptr_t)lds_ptr;
  const unsigned long long gaddr = (unsigned long long)(uintptr_t)gptr;
  asm volatile("global_load_async_to_lds_b128 %0, %1, off"
               :: "v"(laddr), "v"(gaddr)
               : "memory");
}

template <int N>
__device__ __forceinline__ void wait_asynccnt() {
  asm volatile("s_wait_asynccnt %0" :: "i"(N) : "memory");
}

// ---- WMMA fragment loaders (contiguous k = 16*(lane>>4) + elem, wave32) --------------

// A fragment: 16(M) x 32(K) bf16, row-major source with row stride ld.
__device__ __forceinline__ bf16x16 load_a_frag(const bf16* base, int ld) {
  const int lane = lane_id();
  const int row  = lane & 15;
  const int kh   = lane >> 4;
  bf16x16 a;
#pragma unroll
  for (int v = 0; v < 8; ++v) {
    const int k = (kh << 4) + (v << 1);
    a[2 * v]     = base[row * ld + k];
    a[2 * v + 1] = base[row * ld + k + 1];
  }
  return a;
}

// B fragment: 32(K) x 16(N) bf16, SOURCE stores B^T: src[n][k], stride ld.
__device__ __forceinline__ bf16x16 load_bt_frag(const bf16* base, int ld) {
  const int lane = lane_id();
  const int n  = lane & 15;
  const int kh = lane >> 4;
  bf16x16 b;
#pragma unroll
  for (int v = 0; v < 8; ++v) {
    const int k = (kh << 4) + (v << 1);
    b[2 * v]     = base[n * ld + k];
    b[2 * v + 1] = base[n * ld + k + 1];
  }
  return b;
}

// B fragment: 32(K) x 16(N) bf16, natural K-major source src[k][n], stride ld.
__device__ __forceinline__ bf16x16 load_b_frag(const bf16* base, int ld) {
  const int lane = lane_id();
  const int n  = lane & 15;
  const int kh = lane >> 4;
  bf16x16 b;
#pragma unroll
  for (int v = 0; v < 8; ++v) {
    const int k = (kh << 4) + (v << 1);
    b[2 * v]     = base[k * ld + n];
    b[2 * v + 1] = base[(k + 1) * ld + n];
  }
  return b;
}

__device__ __forceinline__ f32x8 wmma_bf16(bf16x16 a, bf16x16 b, f32x8 c) {
  return __builtin_amdgcn_wmma_f32_16x16x32_bf16(false, a, false, b,
                                                 (short)0, c, false, false);
}

// ---- fp32 -> bf16 cast ------------------------------------------------------

__global__ __launch_bounds__(256) void cast_f32_to_bf16(
    const float* __restrict__ in, bf16* __restrict__ out, int n) {
  int i = (int)(blockIdx.x * blockDim.x + threadIdx.x) * 4;
  if (i + 3 < n) {
    float4 f = *reinterpret_cast<const float4*>(in + i);
    out[i + 0] = (bf16)f.x;
    out[i + 1] = (bf16)f.y;
    out[i + 2] = (bf16)f.z;
    out[i + 3] = (bf16)f.w;
  }
}

// ---- shared staging helper for the two projection GEMMs ---------------------
// Issues 3 async b128 copies per thread: 128x32 A tile + 64x32 B tile,
// LDS rows padded to 40 bf16 (80B, multiple of 16B) -> conflict-free frag
// reads and aligned b128 writes.

__device__ __forceinline__ void stage_tiles_async(
    const bf16* __restrict__ X, const bf16* __restrict__ W,
    int m0, int n0, int k0, int t, bf16* As, bf16* Bs) {
#pragma unroll
  for (int i = 0; i < 2; ++i) {
    const int idx = t + i * 256;
    const int r = idx >> 2, c = (idx & 3) * 8;
    async_copy_b128(&As[r * 40 + c], &X[(m0 + r) * NE + k0 + c]);
  }
  {
    const int r = t >> 2, c = (t & 3) * 8;
    async_copy_b128(&Bs[r * 40 + c], &W[(n0 + r) * NE + k0 + c]);
  }
}

// ---- QKV projection: qkv = x @ w_qkv^T + b, scattered to [B,H,S,D] bf16 -----
// Block tile 128(M) x 64(N), K-step 32, double-buffered async LDS staging.
// 8 waves as 4x2, each wave computes a 32x32 output tile.

__global__ __launch_bounds__(256) void qkv_gemm(
    const bf16* __restrict__ X, const bf16* __restrict__ W,
    const float* __restrict__ bias,
    bf16* __restrict__ Qo, bf16* __restrict__ Ko, bf16* __restrict__ Vo) {
  __shared__ __align__(16) bf16 As[2][128 * 40];
  __shared__ __align__(16) bf16 Bs[2][64 * 40];

  const int t  = (int)threadIdx.x;
  const int m0 = (int)blockIdx.y * 128;
  const int n0 = (int)blockIdx.x * 64;
  const int wid = wave_id();
  const int wm = (wid & 3) * 32;
  const int wn = (wid >> 2) * 32;

  f32x8 acc[2][2] = {};

  stage_tiles_async(X, W, m0, n0, 0, t, As[0], Bs[0]);

  for (int k0 = 0; k0 < NE; k0 += 32) {
    const int cur = (k0 >> 5) & 1;
    if (k0 + 32 < NE) {
      stage_tiles_async(X, W, m0, n0, k0 + 32, t, As[cur ^ 1], Bs[cur ^ 1]);
      wait_asynccnt<3>();   // current slab's 3 chunks have landed in LDS
    } else {
      wait_asynccnt<0>();
    }
    __syncthreads();

    bf16x16 a0 = load_a_frag(&As[cur][wm * 40], 40);
    bf16x16 a1 = load_a_frag(&As[cur][(wm + 16) * 40], 40);
    bf16x16 b0 = load_bt_frag(&Bs[cur][wn * 40], 40);
    bf16x16 b1 = load_bt_frag(&Bs[cur][(wn + 16) * 40], 40);
    acc[0][0] = wmma_bf16(a0, b0, acc[0][0]);
    acc[0][1] = wmma_bf16(a0, b1, acc[0][1]);
    acc[1][0] = wmma_bf16(a1, b0, acc[1][0]);
    acc[1][1] = wmma_bf16(a1, b1, acc[1][1]);
    __syncthreads();
  }

  // Epilogue: bias, fold 1/sqrt(E)=1/32 into Q, scatter to [B,H,S,D] bf16.
  const int lane  = lane_id();
  const int ncol  = lane & 15;
  const int rbase = (lane >> 4) << 3;
#pragma unroll
  for (int i = 0; i < 2; ++i) {
#pragma unroll
    for (int j = 0; j < 2; ++j) {
      const int n = n0 + wn + j * 16 + ncol;
      const float bv    = bias[n];
      const int   which = n >> 10;          // 0=q 1=k 2=v
      const int   e     = n & 1023;
      const int   h     = e >> 6;
      const int   d     = e & 63;
      const float scale = (which == 0) ? 0.03125f : 1.0f;
      bf16* dst = (which == 0) ? Qo : (which == 1) ? Ko : Vo;
#pragma unroll
      for (int v = 0; v < 8; ++v) {
        const int m = m0 + wm + i * 16 + rbase + v;
        const int b = m >> 11;              // /2048
        const int s = m & 2047;
        const float val = (acc[i][j][v] + bv) * scale;
        dst[((b * NH + h) * NS + s) * ND + d] = (bf16)val;
      }
    }
  }
}

// ---- Fused attention --------------------------------------------------------
// One block = one (batch, 16-query tile) and ALL 16 heads (softmax couples
// heads). 8 waves, wave w owns heads 2w, 2w+1. Stream k in 32-wide tiles:
//   scores (QK^T, D=64 -> two 16x16x32 WMMAs) -> LDS
//   softmax across the 16 heads per (q,k)       (no k-rescale needed!)
//   ctx += P @ V (four 16x16x32 WMMAs per head)

__global__ __launch_bounds__(256) void attn_fused(
    const bf16* __restrict__ Qm, const bf16* __restrict__ Km,
    const bf16* __restrict__ Vm, bf16* __restrict__ CTX) {
  __shared__ float sbuf[16 * 32 * 17];   // [(q*32+k)*17 + h], pad 16->17
  __shared__ bf16  pbuf[16 * 16 * 36];   // [(h*16+q)*36 + k], pad 32->36

  const int b   = (int)blockIdx.y;
  const int q0  = (int)blockIdx.x * 16;
  const int wid = wave_id();
  const int lane = lane_id();
  const int h0  = wid * 2;

  // Preload Q A-fragments (16 q-rows x 64 d) per owned head, already * 1/32.
  bf16x16 qa[2][2];
#pragma unroll
  for (int hh = 0; hh < 2; ++hh) {
    const bf16* qp = Qm + ((size_t)(b * NH + h0 + hh) * NS + q0) * ND;
    qa[hh][0] = load_a_frag(qp, ND);
    qa[hh][1] = load_a_frag(qp + 32, ND);
  }

  f32x8 cacc[2][4] = {};

  for (int kt = 0; kt < NS; kt += 32) {
    // ---- scores for this k-tile
#pragma unroll
    for (int hh = 0; hh < 2; ++hh) {
      const int h = h0 + hh;
      const bf16* kp = Km + ((size_t)(b * NH + h) * NS + kt) * ND;
#pragma unroll
      for (int kf = 0; kf < 2; ++kf) {
        bf16x16 kb0 = load_bt_frag(kp + kf * 16 * ND, ND);       // d 0..31
        bf16x16 kb1 = load_bt_frag(kp + kf * 16 * ND + 32, ND);  // d 32..63
        f32x8 s = {};
        s = wmma_bf16(qa[hh][0], kb0, s);
        s = wmma_bf16(qa[hh][1], kb1, s);
        const int kcol  = kf * 16 + (lane & 15);
        const int rbase = (lane >> 4) << 3;
#pragma unroll
        for (int v = 0; v < 8; ++v) {
          const int qr = rbase + v;
          sbuf[(qr * 32 + kcol) * 17 + h] = s[v];
        }
      }
    }
    __syncthreads();

    // ---- softmax over heads: 512 (q,k) sites, 2 per thread
#pragma unroll
    for (int it = 0; it < 2; ++it) {
      const int p  = (int)threadIdx.x + it * 256;
      const int qi = p >> 5;
      const int ki = p & 31;
      const float* sp = &sbuf[(qi * 32 + ki) * 17];
      float mx = sp[0];
#pragma unroll
      for (int h = 1; h < NH; ++h) mx = fmaxf(mx, sp[h]);
      float ev[NH];
      float sum = 0.f;
#pragma unroll
      for (int h = 0; h < NH; ++h) { ev[h] = __expf(sp[h] - mx); sum += ev[h]; }
      const float inv = 1.0f / sum;
#pragma unroll
      for (int h = 0; h < NH; ++h)
        pbuf[(h * 16 + qi) * 36 + ki] = (bf16)(ev[h] * inv);
    }
    __syncthreads();

    // ---- ctx += P @ V
#pragma unroll
    for (int hh = 0; hh < 2; ++hh) {
      const int h = h0 + hh;
      bf16x16 pa = load_a_frag(&pbuf[(h * 16) * 36], 36);
      const bf16* vp = Vm + ((size_t)(b * NH + h) * NS + kt) * ND;
#pragma unroll
      for (int j = 0; j < 4; ++j) {
        bf16x16 vb = load_b_frag(vp + j * 16, ND);
        cacc[hh][j] = wmma_bf16(pa, vb, cacc[hh][j]);
      }
    }
    __syncthreads();
  }

  // ---- write ctx as [B,S,E] bf16 (E = h*64 + d) for the output GEMM
  const int rbase = (lane >> 4) << 3;
#pragma unroll
  for (int hh = 0; hh < 2; ++hh) {
    const int h = h0 + hh;
#pragma unroll
    for (int j = 0; j < 4; ++j) {
      const int d = j * 16 + (lane & 15);
#pragma unroll
      for (int v = 0; v < 8; ++v) {
        const int s = q0 + rbase + v;
        CTX[((size_t)(b * NS + s) * NE) + h * ND + d] = (bf16)cacc[hh][j][v];
      }
    }
  }
}

// ---- Output projection: out = ctx @ w_out^T + b_out (fp32 out) --------------

__global__ __launch_bounds__(256) void out_gemm(
    const bf16* __restrict__ X, const bf16* __restrict__ W,
    const float* __restrict__ bias, float* __restrict__ OUT) {
  __shared__ __align__(16) bf16 As[2][128 * 40];
  __shared__ __align__(16) bf16 Bs[2][64 * 40];

  const int t  = (int)threadIdx.x;
  const int m0 = (int)blockIdx.y * 128;
  const int n0 = (int)blockIdx.x * 64;
  const int wid = wave_id();
  const int wm = (wid & 3) * 32;
  const int wn = (wid >> 2) * 32;

  f32x8 acc[2][2] = {};

  stage_tiles_async(X, W, m0, n0, 0, t, As[0], Bs[0]);

  for (int k0 = 0; k0 < NE; k0 += 32) {
    const int cur = (k0 >> 5) & 1;
    if (k0 + 32 < NE) {
      stage_tiles_async(X, W, m0, n0, k0 + 32, t, As[cur ^ 1], Bs[cur ^ 1]);
      wait_asynccnt<3>();
    } else {
      wait_asynccnt<0>();
    }
    __syncthreads();

    bf16x16 a0 = load_a_frag(&As[cur][wm * 40], 40);
    bf16x16 a1 = load_a_frag(&As[cur][(wm + 16) * 40], 40);
    bf16x16 b0 = load_bt_frag(&Bs[cur][wn * 40], 40);
    bf16x16 b1 = load_bt_frag(&Bs[cur][(wn + 16) * 40], 40);
    acc[0][0] = wmma_bf16(a0, b0, acc[0][0]);
    acc[0][1] = wmma_bf16(a0, b1, acc[0][1]);
    acc[1][0] = wmma_bf16(a1, b0, acc[1][0]);
    acc[1][1] = wmma_bf16(a1, b1, acc[1][1]);
    __syncthreads();
  }

  const int lane  = lane_id();
  const int ncol  = lane & 15;
  const int rbase = (lane >> 4) << 3;
#pragma unroll
  for (int i = 0; i < 2; ++i) {
#pragma unroll
    for (int j = 0; j < 2; ++j) {
      const int n = n0 + wn + j * 16 + ncol;
      const float bv = bias[n];
#pragma unroll
      for (int v = 0; v < 8; ++v) {
        const int m = m0 + wm + i * 16 + rbase + v;
        OUT[(size_t)m * NE + n] = acc[i][j][v] + bv;
      }
    }
  }
}

// ---------------------------------------------------------------------------

extern "C" void kernel_launch(void* const* d_in, const int* in_sizes, int n_in,
                              void* d_out, int out_size, void* d_ws,
                              size_t ws_size, hipStream_t stream) {
  const float* x     = (const float*)d_in[0];
  const float* w_qkv = (const float*)d_in[1];
  const float* b_qkv = (const float*)d_in[2];
  const float* w_out = (const float*)d_in[3];
  const float* b_out = (const float*)d_in[4];
  float* out = (float*)d_out;

  char* ws = (char*)d_ws;
  size_t off = 0;
  auto salloc = [&](size_t bytes) {
    void* p = ws + off;
    off = (off + bytes + 255) & ~(size_t)255;
    return p;
  };
  bf16* x_bf    = (bf16*)salloc((size_t)NM * NE * 2);
  bf16* wqkv_bf = (bf16*)salloc((size_t)N_QKV * NE * 2);
  bf16* wout_bf = (bf16*)salloc((size_t)NE * NE * 2);
  bf16* q_bf    = (bf16*)salloc((size_t)NB * NH * NS * ND * 2);
  bf16* k_bf    = (bf16*)salloc((size_t)NB * NH * NS * ND * 2);
  bf16* v_bf    = (bf16*)salloc((size_t)NB * NH * NS * ND * 2);
  bf16* ctx_bf  = (bf16*)salloc((size_t)NM * NE * 2);

  cast_f32_to_bf16<<<(NM * NE) / 1024, 256, 0, stream>>>(x, x_bf, NM * NE);
  cast_f32_to_bf16<<<(N_QKV * NE) / 1024, 256, 0, stream>>>(w_qkv, wqkv_bf,
                                                            N_QKV * NE);
  cast_f32_to_bf16<<<(NE * NE) / 1024, 256, 0, stream>>>(w_out, wout_bf,
                                                         NE * NE);

  qkv_gemm<<<dim3(N_QKV / 64, NM / 128), 256, 0, stream>>>(
      x_bf, wqkv_bf, b_qkv, q_bf, k_bf, v_bf);

  attn_fused<<<dim3(NS / 16, NB), 256, 0, stream>>>(q_bf, k_bf, v_bf, ctx_bf);

  out_gemm<<<dim3(NE / 64, NM / 128), 256, 0, stream>>>(ctx_bf, wout_bf, b_out,
                                                        out);
}
